// SelfAttention_40338332844754
// MI455X (gfx1250) — matrix operand order repaired and checked by reading, not compiled, not run on hardware
//
#include <hip/hip_runtime.h>
#include <hip/hip_bf16.h>
#include <math.h>
#include <stdint.h>

// ---------------------------------------------------------------------------
// Types matching the gfx1250 WMMA builtin signatures
// ---------------------------------------------------------------------------
typedef __bf16 bf16_t;
typedef __attribute__((ext_vector_type(16))) __bf16 bf16x16;
typedef __attribute__((ext_vector_type(8)))  __bf16 bf16x8;
typedef __attribute__((ext_vector_type(8)))  float  f32x8;
typedef __attribute__((ext_vector_type(4)))  float  f32x4;
typedef __attribute__((ext_vector_type(4)))  unsigned int u32x4;
typedef __attribute__((ext_vector_type(8)))  unsigned int u32x8;

#define S_LEN   2048
#define HID     4096
#define NHEADS  32
#define KVH     8
#define HD      128
#define KVW     (KVH * HD)   // 1024

// ---------------------------------------------------------------------------
// WMMA helpers (CDNA5: wave32, 16x16x32 bf16 -> f32)
// ---------------------------------------------------------------------------
__device__ __forceinline__ f32x8 wmma_bf16(bf16x16 a, bf16x16 b, f32x8 c) {
  return __builtin_amdgcn_wmma_f32_16x16x32_bf16(false, a, false, b, (short)0, c,
                                                 false, false);
}

__device__ __forceinline__ f32x8 f32x8_zero() {
  f32x8 z = {0.f, 0.f, 0.f, 0.f, 0.f, 0.f, 0.f, 0.f};
  return z;
}

// A fragment (16x32): lanes 0-15 row=lane K{0..7,16..23}; lanes 16-31 K{8..15,24..31}
__device__ __forceinline__ bf16x16 ld_a_frag(const bf16_t* p, int ld, int lane) {
  const int row = lane & 15;
  const int k0  = (lane >> 4) * 8;
  bf16x8 lo = *(const bf16x8*)(p + row * ld + k0);
  bf16x8 hi = *(const bf16x8*)(p + row * ld + k0 + 16);
  bf16x16 f;
#pragma unroll
  for (int i = 0; i < 8; i++) { f[i] = lo[i]; f[i + 8] = hi[i]; }
  return f;
}

// B fragment (32x16, KxN), LDS holds tile N-major [n][k] so K is contiguous.
__device__ __forceinline__ bf16x16 ld_b_frag(const bf16_t* p, int ld, int lane) {
  const int n  = lane & 15;
  const int k0 = (lane >> 4) * 8;
  bf16x8 lo = *(const bf16x8*)(p + n * ld + k0);
  bf16x8 hi = *(const bf16x8*)(p + n * ld + k0 + 16);
  bf16x16 f;
#pragma unroll
  for (int i = 0; i < 8; i++) { f[i] = lo[i]; f[i + 8] = hi[i]; }
  return f;
}

// ---------------------------------------------------------------------------
// Tensor Data Mover: 2D tile (bf16) global -> LDS with row padding.
// D# per CDNA5 ISA. Issued via inline asm (portable
// across ROCm7.2 / amdgpu-toolchain builtin arity differences).
//   tile_w       : elements along contiguous dim per row
//   tile_h       : number of rows
//   row_stride   : elements between rows in global memory
//   pad_int_code : DWORDs before pad (0=2,1=4,2=8,3=16,4=32,5=64,...)
//   pad_amt_code : pad DWORDs minus 1
// ---------------------------------------------------------------------------
__device__ __forceinline__ void tdm_load_tile_2d(const void* gptr,
                                                 unsigned lds_byte_off,
                                                 unsigned tile_w, unsigned tile_h,
                                                 unsigned row_stride,
                                                 unsigned pad_int_code,
                                                 unsigned pad_amt_code)
{
  const unsigned long long ga = (unsigned long long)(uintptr_t)gptr;
  u32x4 g0;
  g0[0] = 1u;                                            // count=1 (user D#)
  g0[1] = lds_byte_off;                                  // lds_addr
  g0[2] = (unsigned)(ga & 0xFFFFFFFFu);                  // global_addr[31:0]
  g0[3] = (unsigned)((ga >> 32) & 0x01FFFFFFu) | (2u << 30); // addr[56:32], type=2

  const unsigned long long d0 = row_stride;              // tensor_dim0
  const unsigned long long d1 = (1ull << 20);            // tensor_dim1 (ample)
  const unsigned long long s0 = row_stride;              // tensor_dim0_stride
  u32x8 g1;
  g1[0] = (1u << 16)                                     // data_size = 2B
        | (1u << 20)                                     // pad_enable
        | (pad_int_code << 22) | (pad_amt_code << 25);
  g1[1] = (unsigned)((d0 & 0xFFFFull) << 16);            // dim0[15:0] @ [31:16]
  g1[2] = (unsigned)((d0 >> 16) & 0xFFFFull)             // dim0[31:16]
        | (unsigned)((d1 & 0xFFFFull) << 16);            // dim1[15:0]
  g1[3] = (unsigned)((d1 >> 16) & 0xFFFFull)             // dim1[31:16]
        | (tile_w << 16);                                // tile_dim0
  g1[4] = tile_h;                                        // tile_dim1 (tile_dim2=0)
  g1[5] = (unsigned)(s0 & 0xFFFFFFFFull);                // dim0_stride[31:0]
  g1[6] = (unsigned)((s0 >> 32) & 0xFFFFull);            // dim0_stride[47:32]
  g1[7] = 0u;                                            // dim1_stride unused (2D)
  asm volatile("tensor_load_to_lds %0, %1" :: "s"(g0), "s"(g1) : "memory");
}

__device__ __forceinline__ unsigned lds_off(const void* p) {
  return (unsigned)(uintptr_t)p;   // flat LDS address low 32 bits = LDS byte addr
}

// ---------------------------------------------------------------------------
// Prep kernels (one pass each, far under HBM roofline)
// ---------------------------------------------------------------------------
__global__ void cvt_bf16(const float* __restrict__ in, bf16_t* __restrict__ out,
                         int n)
{
  const int i = blockIdx.x * 256 + threadIdx.x;
  if (i < n) out[i] = (bf16_t)in[i];
}

// W[K][N] f32  ->  Wt[N][K] bf16   (32x32 LDS tile transpose)
__launch_bounds__(256)
__global__ void transpose_cvt(const float* __restrict__ W, bf16_t* __restrict__ Wt,
                              int K, int N)
{
  __shared__ float t[32][33];
  const int n0 = blockIdx.x * 32, k0 = blockIdx.y * 32;
  const int tx = threadIdx.x & 31, ty = threadIdx.x >> 5;   // 32 x 8
#pragma unroll
  for (int i = 0; i < 32; i += 8)
    t[ty + i][tx] = W[(size_t)(k0 + ty + i) * N + (n0 + tx)];
  __syncthreads();
#pragma unroll
  for (int i = 0; i < 32; i += 8)
    Wt[(size_t)(n0 + ty + i) * K + (k0 + tx)] = (bf16_t)t[tx][ty + i];
}

// ---------------------------------------------------------------------------
// GEMM: C[M,N] f32 = A[M,K] (bf16) * Bt[N,K]^T (bf16), TDM double-buffered.
// Tile 128x128x32, 256 threads = 8 waves (4x2), 2x4 WMMA accums per wave.
// LDS tiles padded to stride 40 by the TDM (16 DW rows + 4 DW pad).
// ---------------------------------------------------------------------------
__launch_bounds__(256)
__global__ void gemm_bf16_tdm(const bf16_t* __restrict__ A,
                              const bf16_t* __restrict__ Bt,
                              float* __restrict__ C,
                              int M, int N, int K)
{
  constexpr int BM = 128, BN = 128, BK = 32, LD = 40;
  __shared__ alignas(16) bf16_t sA[2][BM * LD];
  __shared__ alignas(16) bf16_t sB[2][BN * LD];

  const int tid  = threadIdx.x;
  const int wave = tid >> 5;
  const int lane = tid & 31;
  const int m0 = blockIdx.y * BM;
  const int n0 = blockIdx.x * BN;
  const int wm = (wave >> 1) * 32;
  const int wn = (wave & 1) * 64;

  f32x8 acc[2][4];
#pragma unroll
  for (int i = 0; i < 2; i++)
#pragma unroll
    for (int j = 0; j < 4; j++) acc[i][j] = f32x8_zero();

  const bf16_t* gA = A  + (size_t)m0 * K;
  const bf16_t* gB = Bt + (size_t)n0 * K;
  const int nkt = K / BK;

  if (tid < 32) {   // wave 0 drives the TDM
    tdm_load_tile_2d(gA, lds_off(&sA[0][0]), BK, BM, K, 3, 3);
    tdm_load_tile_2d(gB, lds_off(&sB[0][0]), BK, BN, K, 3, 3);
  }

  for (int kt = 0; kt < nkt; kt++) {
    const int cur = kt & 1, nxt = cur ^ 1;
    __syncthreads();                       // buf[nxt] no longer read by anyone
    if (tid < 32) {
      if (kt + 1 < nkt) {
        tdm_load_tile_2d(gA + (size_t)(kt + 1) * BK, lds_off(&sA[nxt][0]),
                         BK, BM, K, 3, 3);
        tdm_load_tile_2d(gB + (size_t)(kt + 1) * BK, lds_off(&sB[nxt][0]),
                         BK, BN, K, 3, 3);
        __builtin_amdgcn_s_wait_tensorcnt(2);  // retire buf[cur] pair
      } else {
        __builtin_amdgcn_s_wait_tensorcnt(0);
      }
    }
    __syncthreads();                       // buf[cur] visible to all waves

    bf16x16 af[2], bfr[4];
#pragma unroll
    for (int i = 0; i < 2; i++)
      af[i] = ld_a_frag(&sA[cur][(wm + i * 16) * LD], LD, lane);
#pragma unroll
    for (int j = 0; j < 4; j++)
      bfr[j] = ld_b_frag(&sB[cur][(wn + j * 16) * LD], LD, lane);
#pragma unroll
    for (int i = 0; i < 2; i++)
#pragma unroll
      for (int j = 0; j < 4; j++)
        acc[i][j] = wmma_bf16(af[i], bfr[j], acc[i][j]);
  }

  const int nl = lane & 15;
  const int mh = (lane >> 4) * 8;
#pragma unroll
  for (int i = 0; i < 2; i++)
#pragma unroll
    for (int j = 0; j < 4; j++)
#pragma unroll
      for (int r = 0; r < 8; r++)
        C[(size_t)(m0 + wm + i * 16 + mh + r) * N + (n0 + wn + j * 16 + nl)] =
            acc[i][j][r];
}

// ---------------------------------------------------------------------------
// RoPE + fp32->bf16 (reference: inv_freq[i] = 10000^(i/128), i = 0..63)
// ---------------------------------------------------------------------------
__global__ void rope_cvt(const float* __restrict__ in, bf16_t* __restrict__ out,
                         int W)
{
  const int c = blockIdx.x * 256 + threadIdx.x;
  const int s = blockIdx.y;
  if (c >= W) return;
  const int j  = c & (HD - 1);
  const int jj = j & 63;
  const float freq = __powf(10000.f, (float)jj * (1.f / 128.f));
  float sn, cs;
  __sincosf((float)s * freq, &sn, &cs);
  const float* row = in + (size_t)s * W + (c & ~(HD - 1));
  const float a = row[jj];
  const float b = row[jj + 64];
  const float r = (j < 64) ? (a * cs + b * sn) : (a * sn - b * cs);
  out[(size_t)s * W + c] = (bf16_t)r;
}

// ---------------------------------------------------------------------------
// Flash attention: grid (S/128, 32 heads), 256 threads = 8 waves, key tiles
// of 64, online softmax, bf16 WMMA, TDM-fed Q/K tiles, bf16 context out.
// ---------------------------------------------------------------------------
#define ABM 128
#define ABN 64
#define KSTR 136   // [key][dim] stride: 64 DW rows + 4 DW TDM pad
#define PSTR 72    // [dim][key] / P stride
#define REG0 9216  // K tile (64*136=8704) or P (8*16*72=9216)
#define REG1 9216  // V^T tile (128*72)

__launch_bounds__(256)
__global__ void attn_kernel(const bf16_t* __restrict__ Q,   // (S,4096) roped bf16
                            const bf16_t* __restrict__ Kc,  // (S,1024) roped bf16
                            const bf16_t* __restrict__ Vc,  // (S,1024) bf16
                            bf16_t* __restrict__ O)         // (S,4096) bf16
{
  __shared__ alignas(16) bf16_t smem[REG0 + REG1];   // 36 KB
  bf16_t* sKP = smem;          // K tile [64][KSTR]  (reused as P [wave][16][PSTR])
  bf16_t* sV  = smem + REG0;   // V^T tile [128][PSTR]

  const int tid   = threadIdx.x;
  const int wave  = tid >> 5;
  const int lane  = tid & 31;
  const int qbase = blockIdx.x * ABM;
  const int head  = blockIdx.y;
  const int qcol  = head * HD;
  const int kcol  = (head >> 2) * HD;   // GQA: 4 query heads per kv head

  // ---- stage Q tile (128x128, TDM, padded to KSTR) then keep frags in regs
  if (tid < 32) {
    tdm_load_tile_2d(Q + (size_t)qbase * HID + qcol, lds_off(smem),
                     HD, ABM, HID, 5, 3);
    __builtin_amdgcn_s_wait_tensorcnt(0);
  }
  __syncthreads();
  bf16x16 qfrag[4];
#pragma unroll
  for (int kc = 0; kc < 4; kc++)
    qfrag[kc] = ld_a_frag(smem + (wave * 16) * KSTR + kc * 32, KSTR, lane);
  __syncthreads();

  f32x8 oacc[8];
#pragma unroll
  for (int t = 0; t < 8; t++) oacc[t] = f32x8_zero();
  float mrow[8], lrow[8];
#pragma unroll
  for (int r = 0; r < 8; r++) { mrow[r] = -1e30f; lrow[r] = 0.f; }

  const float scl = 0.08838834764831845f;   // 1/sqrt(128)
  const int nkt = (qbase + ABM) / ABN;      // causal

  for (int kt = 0; kt < nkt; kt++) {
    const int kb0 = kt * ABN;
    // ---- K tile via TDM; V^T staged manually (transpose); overlap the two
    if (tid < 32)
      tdm_load_tile_2d(Kc + (size_t)kb0 * KVW + kcol, lds_off(sKP),
                       HD, ABN, KVW, 5, 3);
    {
      const int r = tid >> 2, c = (tid & 3) * 32;
      const bf16_t* gv = Vc + (size_t)(kb0 + r) * KVW + kcol + c;
      if (kt + 1 < nkt)
        __builtin_prefetch(gv + (size_t)ABN * KVW, 0, 1);
#pragma unroll
      for (int i = 0; i < 32; i++)
        sV[(c + i) * PSTR + r] = gv[i];
    }
    if (tid < 32) __builtin_amdgcn_s_wait_tensorcnt(0);
    __syncthreads();

    // ---- S = Q K^T
    f32x8 sacc[4];
#pragma unroll
    for (int tn = 0; tn < 4; tn++) sacc[tn] = f32x8_zero();
#pragma unroll
    for (int kc = 0; kc < 4; kc++)
#pragma unroll
      for (int tn = 0; tn < 4; tn++) {
        bf16x16 bfrag = ld_b_frag(sKP + (tn * 16) * KSTR + kc * 32, KSTR, lane);
        sacc[tn] = wmma_bf16(qfrag[kc], bfrag, sacc[tn]);
      }
    __syncthreads();   // all waves done with sK before P overwrites it

    // ---- online softmax (row = r + 8*(lane>=16) within wave's 16 rows)
    float mcur[8];
#pragma unroll
    for (int r = 0; r < 8; r++) {
      const int row = qbase + wave * 16 + r + 8 * (lane >> 4);
      float mx = -1e30f;
#pragma unroll
      for (int tn = 0; tn < 4; tn++) {
        const int col = kb0 + tn * 16 + (lane & 15);
        float v = sacc[tn][r] * scl;
        if (col > row) v = -1e30f;            // causal mask
        sacc[tn][r] = v;
        mx = fmaxf(mx, v);
      }
#pragma unroll
      for (int off = 1; off < 16; off <<= 1)
        mx = fmaxf(mx, __shfl_xor(mx, off, 32));
      mcur[r] = mx;
    }

    bf16_t* sPw = sKP + wave * (16 * PSTR);
#pragma unroll
    for (int r = 0; r < 8; r++) {
      const float mnew = fmaxf(mrow[r], mcur[r]);
      const float resc = __expf(mrow[r] - mnew);
      mrow[r] = mnew;
      float rsum = 0.f;
#pragma unroll
      for (int tn = 0; tn < 4; tn++) {
        const float p = __expf(sacc[tn][r] - mnew);
        rsum += p;
        sPw[(r + 8 * (lane >> 4)) * PSTR + tn * 16 + (lane & 15)] = (bf16_t)p;
      }
#pragma unroll
      for (int off = 1; off < 16; off <<= 1)
        rsum += __shfl_xor(rsum, off, 32);
      lrow[r] = lrow[r] * resc + rsum;
#pragma unroll
      for (int tn = 0; tn < 8; tn++) oacc[tn][r] *= resc;
    }

    // ---- O += P V  (per-wave P slice; in-wave DS ordering suffices)
#pragma unroll
    for (int kc = 0; kc < 2; kc++) {
      bf16x16 pfrag = ld_a_frag(sPw + kc * 32, PSTR, lane);
#pragma unroll
      for (int tn = 0; tn < 8; tn++) {
        bf16x16 vfrag = ld_b_frag(sV + (tn * 16) * PSTR + kc * 32, PSTR, lane);
        oacc[tn] = wmma_bf16(pfrag, vfrag, oacc[tn]);
      }
    }
    __syncthreads();   // before next iteration's TDM/copies overwrite LDS
  }

  // ---- epilogue: normalize, emit bf16 context for the o@Wo GEMM
#pragma unroll
  for (int r = 0; r < 8; r++) {
    const float inv = 1.f / lrow[r];
    const int row = qbase + wave * 16 + r + 8 * (lane >> 4);
#pragma unroll
    for (int tn = 0; tn < 8; tn++)
      O[(size_t)row * HID + qcol + tn * 16 + (lane & 15)] =
          (bf16_t)(oacc[tn][r] * inv);
  }
}

// ---------------------------------------------------------------------------
// Host launcher
// ---------------------------------------------------------------------------
extern "C" void kernel_launch(void* const* d_in, const int* in_sizes, int n_in,
                              void* d_out, int out_size, void* d_ws, size_t ws_size,
                              hipStream_t stream)
{
  (void)in_sizes; (void)n_in; (void)out_size; (void)ws_size;
  const float* x  = (const float*)d_in[0];
  const float* Wq = (const float*)d_in[1];
  const float* Wk = (const float*)d_in[2];
  const float* Wv = (const float*)d_in[3];
  const float* Wo = (const float*)d_in[4];
  float* out = (float*)d_out;

  // Workspace layout (~184 MB)
  bf16_t* xb  = (bf16_t*)d_ws;                         // S*H
  bf16_t* Wqt = xb  + (size_t)S_LEN * HID;             // H*H   [N][K]
  bf16_t* Wkt = Wqt + (size_t)HID * HID;               // KVW*H [N][K]
  bf16_t* Wvt = Wkt + (size_t)KVW * HID;
  bf16_t* Wot = Wvt + (size_t)KVW * HID;               // H*H   [N][K]
  float*  qf  = (float*)(Wot + (size_t)HID * HID);     // S*H   f32
  float*  kf  = qf + (size_t)S_LEN * HID;              // S*KVW f32
  float*  vf  = kf + (size_t)S_LEN * KVW;              // S*KVW f32
  bf16_t* qb  = (bf16_t*)(vf + (size_t)S_LEN * KVW);   // S*H   (roped)
  bf16_t* kb  = qb + (size_t)S_LEN * HID;              // S*KVW (roped)
  bf16_t* vb  = kb + (size_t)S_LEN * KVW;              // S*KVW
  bf16_t* ob  = vb + (size_t)S_LEN * KVW;              // S*H   context

  dim3 blk(256);
  // prep: bf16 x, bf16 transposed weights
  cvt_bf16<<<dim3((S_LEN * HID) / 256), blk, 0, stream>>>(x, xb, S_LEN * HID);
  transpose_cvt<<<dim3(HID / 32, HID / 32), blk, 0, stream>>>(Wq, Wqt, HID, HID);
  transpose_cvt<<<dim3(KVW / 32, HID / 32), blk, 0, stream>>>(Wk, Wkt, HID, KVW);
  transpose_cvt<<<dim3(KVW / 32, HID / 32), blk, 0, stream>>>(Wv, Wvt, HID, KVW);
  transpose_cvt<<<dim3(HID / 32, HID / 32), blk, 0, stream>>>(Wo, Wot, HID, HID);

  // projections (TDM-fed bf16 WMMA)
  gemm_bf16_tdm<<<dim3(HID / 128, S_LEN / 128), blk, 0, stream>>>(
      xb, Wqt, qf, S_LEN, HID, HID);
  gemm_bf16_tdm<<<dim3(KVW / 128, S_LEN / 128), blk, 0, stream>>>(
      xb, Wkt, kf, S_LEN, KVW, HID);
  gemm_bf16_tdm<<<dim3(KVW / 128, S_LEN / 128), blk, 0, stream>>>(
      xb, Wvt, vf, S_LEN, KVW, HID);

  // rotary + dtype for attention operands
  rope_cvt<<<dim3(HID / 256, S_LEN), blk, 0, stream>>>(qf, qb, HID);
  rope_cvt<<<dim3(KVW / 256, S_LEN), blk, 0, stream>>>(kf, kb, KVW);
  cvt_bf16<<<dim3((S_LEN * KVW) / 256), blk, 0, stream>>>(vf, vb, S_LEN * KVW);

  // flash attention (bf16 context out)
  attn_kernel<<<dim3(S_LEN / ABM, NHEADS), blk, 0, stream>>>(qb, kb, vb, ob);

  // output projection
  gemm_bf16_tdm<<<dim3(HID / 128, S_LEN / 128), blk, 0, stream>>>(
      ob, Wot, out, S_LEN, HID, HID);
}
